// RWKV7RNN_37735582662869
// MI455X (gfx1250) — matrix operand order repaired and checked by reading, not compiled, not run on hardware
//
#include <hip/hip_runtime.h>
#include <cstdint>
#include <cstddef>

// ---------------------------------------------------------------------------
// RWKV7 forward, MI455X (gfx1250, wave32).
//   - Big GEMMs: bf16 WMMA (v_wmma_f32_16x16x32_bf16), fp32 accumulate.
//   - Weight/activation tiles streamed by the Tensor Data Mover
//     (tensor_load_to_lds) into a ping-pong LDS double buffer, synchronized
//     with s_wait_tensorcnt(2) so DMA overlaps the WMMA pipeline.
//   - Delta-rule state update factored to rank-1 form (pure VALU, tiny).
// ---------------------------------------------------------------------------

#define LNUM 12
#define BB   128
#define CC   1024
#define HH   16
#define KK   64
#define KDIM 4096
#define DV_  32
#define DA_  64
#define DD_  64
#define DG_  128

typedef __attribute__((ext_vector_type(16))) __bf16 v16bf;
typedef __attribute__((ext_vector_type(8)))  float  v8f;
typedef __attribute__((ext_vector_type(4)))  unsigned int u32x4;
typedef __attribute__((ext_vector_type(8)))  int    i32x8;
typedef __attribute__((ext_vector_type(4)))  int    i32x4;

#if defined(__gfx1250__) && __has_builtin(__builtin_amdgcn_tensor_load_to_lds)
#define USE_TDM 1
#else
#define USE_TDM 0
#endif

__device__ __forceinline__ float wave_sum(float x) {
  x += __shfl_xor(x, 16, 32);
  x += __shfl_xor(x, 8, 32);
  x += __shfl_xor(x, 4, 32);
  x += __shfl_xor(x, 2, 32);
  x += __shfl_xor(x, 1, 32);
  return x;
}

__device__ __forceinline__ float sigmoidf_(float x) { return 1.f / (1.f + __expf(-x)); }

// --------------------------- TDM 2D tile load ------------------------------
// data_size fixed 4 bytes. tile_w elems contiguous (dim0), tile_h rows (dim1),
// row stride = stride elems. Descriptor bit layout per CDNA5 ISA ch. 8.
__device__ __forceinline__ void tdm_load_2d(unsigned lds_off, const void* gptr,
                                            unsigned tile_w, unsigned tile_h,
                                            unsigned tensor_w, unsigned tensor_h,
                                            unsigned stride_elems) {
#if USE_TDM
  unsigned long long ga = (unsigned long long)(uintptr_t)gptr;
  u32x4 g0;
  g0[0] = 1u;                                   // count=1, user descriptor
  g0[1] = lds_off;                              // lds_addr (bytes)
  g0[2] = (unsigned)(ga & 0xffffffffull);       // global_addr[31:0]
  g0[3] = (unsigned)((ga >> 32) & 0x1ffffffull) | (2u << 30); // [56:32] | type=2
  i32x8 g1;
  g1[0] = (int)(2u << 16);                      // wg_mask=0, data_size=2 (4B)
  g1[1] = (int)((tensor_w & 0xffffu) << 16);    // tensor_dim0[15:0]
  g1[2] = (int)((tensor_w >> 16) & 0xffffu) | (int)((tensor_h & 0xffffu) << 16);
  g1[3] = (int)((tensor_h >> 16) & 0xffffu) | (int)((tile_w & 0xffffu) << 16);
  g1[4] = (int)(tile_h & 0xffffu);              // tile_dim1 (tile_dim2=0)
  g1[5] = (int)stride_elems;                    // tensor_dim0_stride[31:0]
  g1[6] = 0;
  g1[7] = 0;
  i32x4 gz4 = {0, 0, 0, 0};
  i32x8 gz8 = {0, 0, 0, 0, 0, 0, 0, 0};
  __builtin_amdgcn_tensor_load_to_lds(g0, g1, gz4, gz4, gz8, 0);
#else
  (void)lds_off; (void)gptr; (void)tile_w; (void)tile_h;
  (void)tensor_w; (void)tensor_h; (void)stride_elems;
#endif
}

// --------------------------- WMMA GEMM (M=128) -----------------------------
// out(128,Nt) = epi( A(128,Kt) @ W(Kt,Nt) [+ res] ).  EPI: 0 store, 1 +res, 2 relu^2
template <int EPI>
__global__ __launch_bounds__(256) void gemm128_kernel(
    const float* __restrict__ A, const float* __restrict__ W,
    const float* __restrict__ res, float* __restrict__ out, int Kt, int Nt) {
  __shared__ float ldsA[2][128 * 32];   // [m][k] ping-pong
  __shared__ float ldsW[2][32 * 128];   // [k][n] ping-pong
  const int tid  = threadIdx.x;
  const int wave = tid >> 5, lane = tid & 31;
  const int l16  = lane & 15;
  const int hi   = lane >> 4;                 // 0|1
  const int wm   = wave & 3, wn = wave >> 2;  // 4x2 wave grid -> 32x64 per wave
  const int n0   = blockIdx.x * 128;
  const int koffA = hi ? 8 : 0;
  const int koffB = hi ? 8 : 0;
  const int nsteps = Kt >> 5;
  v8f acc[2][4] = {};

#if USE_TDM
  if (wave == 0) {   // preload tile 0 into buffer 0
    tdm_load_2d((unsigned)(uintptr_t)&ldsA[0][0], A, 32, 128,
                (unsigned)Kt, 128u, (unsigned)Kt);
    tdm_load_2d((unsigned)(uintptr_t)&ldsW[0][0], W + n0, 128, 32,
                (unsigned)Nt, (unsigned)Kt, (unsigned)Nt);
  }
#endif

  for (int s = 0; s < nsteps; ++s) {
    const int cur = s & 1;
#if USE_TDM
    if (wave == 0) {
      if (s + 1 < nsteps) {
        // stream next tile into the other buffer while this one computes
        const int k1 = (s + 1) << 5;
        tdm_load_2d((unsigned)(uintptr_t)&ldsA[cur ^ 1][0], A + k1, 32, 128,
                    (unsigned)Kt, 128u, (unsigned)Kt);
        tdm_load_2d((unsigned)(uintptr_t)&ldsW[cur ^ 1][0],
                    W + (size_t)k1 * Nt + n0, 128, 32,
                    (unsigned)Nt, (unsigned)Kt, (unsigned)Nt);
        __builtin_amdgcn_s_wait_tensorcnt(2);  // tile s resident (in-order)
      } else {
        __builtin_amdgcn_s_wait_tensorcnt(0);
      }
    }
#else
    {
      const int k0 = s << 5;
      int row = tid >> 1, kp = (tid & 1) * 16;
      const float* gA = A + (size_t)row * Kt + k0 + kp;
      float* dA = &ldsA[cur][row * 32 + kp];
#pragma unroll
      for (int e = 0; e < 16; ++e) dA[e] = gA[e];
      int kr = tid >> 3, np = (tid & 7) * 16;
      const float* gW = W + (size_t)(k0 + kr) * Nt + n0 + np;
      float* dW = &ldsW[cur][kr * 128 + np];
#pragma unroll
      for (int e = 0; e < 16; ++e) dW[e] = gW[e];
    }
#endif
    __syncthreads();

    // Fragment assembly per ISA VGPR layouts (convert f32 -> bf16 here).
    const float* bufA = &ldsA[cur][0];
    const float* bufW = &ldsW[cur][0];
    v16bf af[2], bfr[4];
#pragma unroll
    for (int i = 0; i < 2; ++i) {
      const float* p = &bufA[((wm * 2 + i) * 16 + l16) * 32 + koffA];
#pragma unroll
      for (int e = 0; e < 8; ++e) {
        af[i][e]     = (__bf16)p[e];        // K = koffA + e
        af[i][8 + e] = (__bf16)p[16 + e];   // K = koffA + 16 + e
      }
    }
#pragma unroll
    for (int j = 0; j < 4; ++j) {
      const int col = wn * 64 + j * 16 + l16;
#pragma unroll
      for (int e = 0; e < 16; ++e)
        bfr[j][e] = (__bf16)bufW[(koffB + e + (e >> 3) * 8) * 128 + col];  // K = koffB + e (+8 for e>=8)
    }
#pragma unroll
    for (int i = 0; i < 2; ++i)
#pragma unroll
      for (int j = 0; j < 4; ++j)
        acc[i][j] = __builtin_amdgcn_wmma_f32_16x16x32_bf16(
            false, af[i], false, bfr[j], (short)0, acc[i][j], false, false);
    __syncthreads();
  }

#pragma unroll
  for (int i = 0; i < 2; ++i)
#pragma unroll
    for (int j = 0; j < 4; ++j) {
      const int col = n0 + wn * 64 + j * 16 + l16;
#pragma unroll
      for (int e = 0; e < 8; ++e) {
        const int row = (wm * 2 + i) * 16 + hi * 8 + e;  // C/D layout
        float v = acc[i][j][e];
        if (EPI == 1) v += res[(size_t)row * Nt + col];
        if (EPI == 2) { v = fmaxf(v, 0.f); v = v * v; }
        out[(size_t)row * Nt + col] = v;
      }
    }
}

// --------------------- LayerNorm + 8-way token-shift lerp ------------------
__global__ __launch_bounds__(256) void ln_mix8_kernel(
    const float* __restrict__ x, const float* __restrict__ xs,
    const float* __restrict__ w, const float* __restrict__ b,
    const float* __restrict__ lr8, float* __restrict__ mixed) {
  const int bi = blockIdx.x, tid = threadIdx.x;
  const float* xb = x + (size_t)bi * CC;
  float vals[4], s = 0.f, s2 = 0.f;
#pragma unroll
  for (int i = 0; i < 4; ++i) {
    float v = xb[tid + i * 256];
    vals[i] = v; s += v; s2 += v * v;
  }
  s = wave_sum(s); s2 = wave_sum(s2);
  __shared__ float red[16];
  if ((tid & 31) == 0) { red[tid >> 5] = s; red[8 + (tid >> 5)] = s2; }
  __syncthreads();
  if (tid == 0) {
    float a = 0.f, c = 0.f;
    for (int i = 0; i < 8; ++i) { a += red[i]; c += red[8 + i]; }
    red[0] = a; red[8] = c;
  }
  __syncthreads();
  const float mean = red[0] * (1.f / CC);
  const float var  = red[8] * (1.f / CC) - mean * mean;
  const float inv  = rsqrtf(var + 1e-5f);
#pragma unroll
  for (int i = 0; i < 4; ++i) {
    const int c = tid + i * 256;
    const float xl  = (vals[i] - mean) * inv * w[c] + b[c];
    const float xsv = xs[(size_t)bi * CC + c];
#pragma unroll
    for (int j = 0; j < 8; ++j)
      mixed[((size_t)j * BB + bi) * CC + c] = xl + lr8[j * CC + c] * (xsv - xl);
  }
}

__global__ __launch_bounds__(256) void ln_mix1_kernel(
    const float* __restrict__ x, const float* __restrict__ xs,
    const float* __restrict__ w, const float* __restrict__ b,
    const float* __restrict__ lk, float* __restrict__ outm) {
  const int bi = blockIdx.x, tid = threadIdx.x;
  const float* xb = x + (size_t)bi * CC;
  float vals[4], s = 0.f, s2 = 0.f;
#pragma unroll
  for (int i = 0; i < 4; ++i) {
    float v = xb[tid + i * 256];
    vals[i] = v; s += v; s2 += v * v;
  }
  s = wave_sum(s); s2 = wave_sum(s2);
  __shared__ float red[16];
  if ((tid & 31) == 0) { red[tid >> 5] = s; red[8 + (tid >> 5)] = s2; }
  __syncthreads();
  if (tid == 0) {
    float a = 0.f, c = 0.f;
    for (int i = 0; i < 8; ++i) { a += red[i]; c += red[8 + i]; }
    red[0] = a; red[8] = c;
  }
  __syncthreads();
  const float mean = red[0] * (1.f / CC);
  const float var  = red[8] * (1.f / CC) - mean * mean;
  const float inv  = rsqrtf(var + 1e-5f);
#pragma unroll
  for (int i = 0; i < 4; ++i) {
    const int c = tid + i * 256;
    const float xl  = (vals[i] - mean) * inv * w[c] + b[c];
    const float xsv = xs[(size_t)bi * CC + c];
    outm[(size_t)bi * CC + c] = xl + lk[c] * (xsv - xl);
  }
}

// ------------------ sigmoid(x @ W^T + b) heads, (B,H) ----------------------
__global__ __launch_bounds__(256) void head_scale_kernel(
    const float* __restrict__ mixed, const float* __restrict__ ksW,
    const float* __restrict__ ksb, const float* __restrict__ vsW,
    const float* __restrict__ vsb, float* __restrict__ ks, float* __restrict__ vs) {
  const int bi = blockIdx.x, tid = threadIdx.x;
  const int g = tid >> 3, l8 = tid & 7;
  const int h = g & 15, sel = g >> 4;
  const float* xin = mixed + ((size_t)(sel ? 7 : 6) * BB + bi) * CC;
  const float* wr  = (sel ? vsW : ksW) + (size_t)h * CC;
  float s = 0.f;
  for (int c = l8; c < CC; c += 8) s += xin[c] * wr[c];
  s += __shfl_xor(s, 1, 32); s += __shfl_xor(s, 2, 32); s += __shfl_xor(s, 4, 32);
  if (l8 == 0) {
    const float bias = sel ? vsb[h] : ksb[h];
    const float r = sigmoidf_(s + bias);
    if (sel) vs[bi * HH + h] = r; else ks[bi * HH + h] = r;
  }
}

// ------------------------------ LoRA stages --------------------------------
template <int ACT>  // 0 none, 1 tanh, 2 sigmoid
__global__ void lora_stage1(const float* __restrict__ X, const float* __restrict__ A,
                            float* __restrict__ out, int D) {
  const int bi = blockIdx.x, d = threadIdx.x;
  if (d >= D) return;
  const float* xb = X + (size_t)bi * CC;
  float s = 0.f;
  for (int c = 0; c < CC; ++c) s += xb[c] * A[(size_t)c * D + d];
  if (ACT == 1) s = tanhf(s);
  if (ACT == 2) s = sigmoidf_(s);
  out[bi * D + d] = s;
}

template <int ACT>  // 0 none, 2 sigmoid
__global__ __launch_bounds__(256) void lora_stage2(
    const float* __restrict__ T, const float* __restrict__ Bm,
    const float* __restrict__ bias, float* __restrict__ out, int D) {
  const int bi = blockIdx.x, tid = threadIdx.x;
  __shared__ float sh[DG_];
  if (tid < D) sh[tid] = T[bi * D + tid];
  __syncthreads();
  for (int c = tid; c < CC; c += 256) {
    float s = bias ? bias[c] : 0.f;
    for (int d = 0; d < D; ++d) s += sh[d] * Bm[(size_t)d * CC + c];
    if (ACT == 2) s = sigmoidf_(s);
    out[(size_t)bi * CC + c] = s;
  }
}

// ------------- factored delta-rule state + GroupNorm + bonus ---------------
// out[i] = sum_j S[i,j]*w[j]*r[j] + (vh[i] - S[i,:]·kk) * (kh·r)
__global__ __launch_bounds__(32) void state_kernel(
    const float* __restrict__ S, const float* __restrict__ r,
    const float* __restrict__ k, const float* __restrict__ vwv,
    const float* __restrict__ vmix, const float* __restrict__ aout,
    const float* __restrict__ dbuf, const float* __restrict__ gbuf,
    const float* __restrict__ ks, const float* __restrict__ vs,
    const float* __restrict__ bon, const float* __restrict__ gnw,
    const float* __restrict__ gnb, float* __restrict__ v0,
    float* __restrict__ y, int first) {
  const int bh = blockIdx.x, bi = bh >> 4, h = bh & 15, lane = threadIdx.x;
  const int cb = bi * CC + h * KK;
  __shared__ float skk[KK], swr[KK];
  float kv[2], vv[2], av[2], wv[2], rv[2], bv[2];
#pragma unroll
  for (int t = 0; t < 2; ++t) {
    const int e = lane + t * 32, c = cb + e;
    kv[t] = k[c];
    const float vw = vwv[c];
    vv[t] = first ? vw : vw + vmix[c] * (v0[c] - vw);
    av[t] = aout[c];
    const float nx = -dbuf[c];
    const float sp = nx > 20.f ? nx : log1pf(__expf(nx));  // softplus(-d)
    wv[t] = __expf(-__expf(-0.5f - sp));
    rv[t] = r[c];
    bv[t] = bon[h * KK + e];
  }
  const float ksc = ks[bi * HH + h], vsc = vs[bi * HH + h];
  float sk = wave_sum(kv[0] * kv[0] + kv[1] * kv[1]);
  float sv = wave_sum(vv[0] * vv[0] + vv[1] * vv[1]);
  const float kn = ksc / fmaxf(sqrtf(sk), 1e-12f);
  const float vn = vsc / fmaxf(sqrtf(sv), 1e-12f);
  float kk2[2], kh2[2], vh2[2], c1p = 0.f, bnp = 0.f;
#pragma unroll
  for (int t = 0; t < 2; ++t) {
    kk2[t] = kn * kv[t];
    kh2[t] = kk2[t] * av[t];
    vh2[t] = vn * vv[t];
    c1p += kh2[t] * rv[t];
    bnp += rv[t] * bv[t] * kh2[t];
  }
  const float c1 = wave_sum(c1p);
  const float bsum = wave_sum(bnp);
  skk[lane] = kk2[0]; skk[lane + 32] = kk2[1];
  swr[lane] = wv[0] * rv[0]; swr[lane + 32] = wv[1] * rv[1];
  __syncthreads();
  float outv[2];
#pragma unroll
  for (int t = 0; t < 2; ++t) {
    const int i = lane + t * 32;
    const float* Srow = S + ((size_t)bh * KK + i) * KK;
    float dot = 0.f, tt = 0.f;
#pragma unroll 8
    for (int j = 0; j < KK; ++j) {
      const float s = Srow[j];
      dot += s * skk[j];
      tt += s * swr[j];
    }
    outv[t] = tt + (vh2[t] - dot) * c1;
  }
  const float mu = wave_sum(outv[0] + outv[1]) * (1.f / KK);
  float vp = (outv[0] - mu) * (outv[0] - mu) + (outv[1] - mu) * (outv[1] - mu);
  const float inv = rsqrtf(wave_sum(vp) * (1.f / KK) + 6.4e-4f);
#pragma unroll
  for (int t = 0; t < 2; ++t) {
    const int e = lane + t * 32, c = cb + e, cc = h * KK + e;
    const float gn = (outv[t] - mu) * inv * gnw[cc] + gnb[cc];
    y[c] = gbuf[c] * (gn + bsum * vh2[t]);
    if (first) v0[c] = vwv[c];
  }
}

// ------------------------------- launcher ----------------------------------
extern "C" void kernel_launch(void* const* d_in, const int* in_sizes, int n_in,
                              void* d_out, int out_size, void* d_ws, size_t ws_size,
                              hipStream_t stream) {
  (void)in_sizes; (void)n_in; (void)out_size; (void)ws_size;
  const float* in_BC   = (const float*)d_in[0];
  const float* xs_t    = (const float*)d_in[1];
  const float* S_all   = (const float*)d_in[2];
  const float* xs_c    = (const float*)d_in[3];
  const float* ln_t_w  = (const float*)d_in[4];
  const float* ln_t_b  = (const float*)d_in[5];
  const float* rkvdag  = (const float*)d_in[6];
  const float* bonus   = (const float*)d_in[7];
  const float* W_r     = (const float*)d_in[8];
  const float* W_k     = (const float*)d_in[9];
  const float* W_v     = (const float*)d_in[10];
  const float* W_o     = (const float*)d_in[11];
  const float* ks_W    = (const float*)d_in[12];
  const float* ks_b    = (const float*)d_in[13];
  const float* vs_W    = (const float*)d_in[14];
  const float* vs_b    = (const float*)d_in[15];
  const float* vA      = (const float*)d_in[16];
  const float* vB      = (const float*)d_in[17];
  const float* v_bias  = (const float*)d_in[18];
  const float* aA      = (const float*)d_in[19];
  const float* aB      = (const float*)d_in[20];
  const float* a_bias  = (const float*)d_in[21];
  const float* dA      = (const float*)d_in[22];
  const float* dB      = (const float*)d_in[23];
  const float* d_bias  = (const float*)d_in[24];
  const float* gA      = (const float*)d_in[25];
  const float* gB      = (const float*)d_in[26];
  const float* gn_w    = (const float*)d_in[27];
  const float* gn_b    = (const float*)d_in[28];
  const float* ln_c_w  = (const float*)d_in[29];
  const float* ln_c_b  = (const float*)d_in[30];
  const float* lerp_k  = (const float*)d_in[31];
  const float* cW_k    = (const float*)d_in[32];
  const float* cW_v    = (const float*)d_in[33];

  const size_t BC = (size_t)BB * CC;
  float* ws     = (float*)d_ws;
  float* x      = ws;
  float* v0     = ws + BC;
  float* mixed  = ws + 2 * BC;    // 8*BC
  float* rbuf   = ws + 10 * BC;
  float* kbuf   = ws + 11 * BC;
  float* vwv    = ws + 12 * BC;
  float* vmix   = ws + 13 * BC;
  float* abuf   = ws + 14 * BC;
  float* dbuf   = ws + 15 * BC;
  float* gbuf   = ws + 16 * BC;
  float* ybuf   = ws + 17 * BC;
  float* xt     = ws + 18 * BC;
  float* mixc   = ws + 19 * BC;
  float* hidden = ws + 20 * BC;   // 4*BC
  float* ksbuf  = ws + 24 * BC;
  float* vsbuf  = ksbuf + BB * HH;
  float* tmp    = vsbuf + BB * HH;

  (void)hipMemcpyAsync(x, in_BC, BC * sizeof(float), hipMemcpyDeviceToDevice, stream);

  for (int l = 0; l < LNUM; ++l) {
    const float* Sl   = S_all + (size_t)l * BB * HH * KK * KK;
    const float* xstl = xs_t + (size_t)l * BC;
    const float* xscl = xs_c + (size_t)l * BC;
    const float* lnw  = ln_t_w + (size_t)l * CC;
    const float* lnb  = ln_t_b + (size_t)l * CC;
    const float* lr8  = rkvdag + (size_t)l * 8 * CC;
    const float* bonl = bonus + (size_t)l * HH * KK;
    const float* Wrl  = W_r + (size_t)l * CC * CC;
    const float* Wkl  = W_k + (size_t)l * CC * CC;
    const float* Wvl  = W_v + (size_t)l * CC * CC;
    const float* Wol  = W_o + (size_t)l * CC * CC;
    const float* ksWl = ks_W + (size_t)l * HH * CC;
    const float* ksbl = ks_b + (size_t)l * HH;
    const float* vsWl = vs_W + (size_t)l * HH * CC;
    const float* vsbl = vs_b + (size_t)l * HH;
    const float* vAl  = vA + (size_t)l * CC * DV_;
    const float* vBl  = vB + (size_t)l * DV_ * CC;
    const float* vbl  = v_bias + (size_t)l * CC;
    const float* aAl  = aA + (size_t)l * CC * DA_;
    const float* aBl  = aB + (size_t)l * DA_ * CC;
    const float* abl  = a_bias + (size_t)l * CC;
    const float* dAl  = dA + (size_t)l * CC * DD_;
    const float* dBl  = dB + (size_t)l * DD_ * CC;
    const float* dbl  = d_bias + (size_t)l * CC;
    const float* gAl  = gA + (size_t)l * CC * DG_;
    const float* gBl  = gB + (size_t)l * DG_ * CC;
    const float* gnwl = gn_w + (size_t)l * CC;
    const float* gnbl = gn_b + (size_t)l * CC;
    const float* lcwl = ln_c_w + (size_t)l * CC;
    const float* lcbl = ln_c_b + (size_t)l * CC;
    const float* lkl  = lerp_k + (size_t)l * CC;
    const float* cWkl = cW_k + (size_t)l * CC * KDIM;
    const float* cWvl = cW_v + (size_t)l * KDIM * CC;

    // ---- time mixer ----
    ln_mix8_kernel<<<BB, 256, 0, stream>>>(x, xstl, lnw, lnb, lr8, mixed);
    head_scale_kernel<<<BB, 256, 0, stream>>>(mixed, ksWl, ksbl, vsWl, vsbl, ksbuf, vsbuf);

    // LoRAs: v(mixed2), a(mixed4), d(mixed3, tanh mid), g(mixed5, sigmoid mid)
    lora_stage1<0><<<BB, DV_, 0, stream>>>(mixed + 2 * BC, vAl, tmp, DV_);
    lora_stage2<2><<<BB, 256, 0, stream>>>(tmp, vBl, vbl, vmix, DV_);
    lora_stage1<0><<<BB, DA_, 0, stream>>>(mixed + 4 * BC, aAl, tmp, DA_);
    lora_stage2<2><<<BB, 256, 0, stream>>>(tmp, aBl, abl, abuf, DA_);
    lora_stage1<1><<<BB, DD_, 0, stream>>>(mixed + 3 * BC, dAl, tmp, DD_);
    lora_stage2<0><<<BB, 256, 0, stream>>>(tmp, dBl, dbl, dbuf, DD_);
    lora_stage1<2><<<BB, DG_, 0, stream>>>(mixed + 5 * BC, gAl, tmp, DG_);
    lora_stage2<0><<<BB, 256, 0, stream>>>(tmp, gBl, nullptr, gbuf, DG_);

    // big GEMMs: r, k, v
    gemm128_kernel<0><<<CC / 128, 256, 0, stream>>>(mixed + 0 * BC, Wrl, nullptr, rbuf, CC, CC);
    gemm128_kernel<0><<<CC / 128, 256, 0, stream>>>(mixed + 1 * BC, Wkl, nullptr, kbuf, CC, CC);
    gemm128_kernel<0><<<CC / 128, 256, 0, stream>>>(mixed + 2 * BC, Wvl, nullptr, vwv, CC, CC);

    // factored delta-rule state + GN + bonus -> gated y
    state_kernel<<<BB * HH, 32, 0, stream>>>(Sl, rbuf, kbuf, vwv, vmix, abuf, dbuf,
                                             gbuf, ksbuf, vsbuf, bonl, gnwl, gnbl,
                                             v0, ybuf, (l == 0) ? 1 : 0);

    // xt = x + y @ Wo
    gemm128_kernel<1><<<CC / 128, 256, 0, stream>>>(ybuf, Wol, x, xt, CC, CC);

    // ---- channel mixer ----
    ln_mix1_kernel<<<BB, 256, 0, stream>>>(xt, xscl, lcwl, lcbl, lkl, mixc);
    gemm128_kernel<2><<<KDIM / 128, 256, 0, stream>>>(mixc, cWkl, nullptr, hidden, CC, KDIM);
    gemm128_kernel<1><<<CC / 128, 256, 0, stream>>>(hidden, cWvl, xt, x, KDIM, CC);
  }

  (void)hipMemcpyAsync(d_out, x, BC * sizeof(float), hipMemcpyDeviceToDevice, stream);
}
